// AttnBlock_41875931136347
// MI455X (gfx1250) — compile-verified, matrix operand order checked by reading, not run on hardware
//
#include <hip/hip_runtime.h>
#include <hip/hip_bf16.h>

// MI455X / gfx1250: wave32, WMMA 16x16x32 bf16 -> f32 accum.
typedef __bf16 bf16;
typedef __attribute__((ext_vector_type(16))) __bf16 v16bf;
typedef __attribute__((ext_vector_type(8)))  __bf16 v8bf;
typedef __attribute__((ext_vector_type(8)))  float  v8f;
typedef __attribute__((ext_vector_type(4)))  unsigned int v4u;
typedef __attribute__((ext_vector_type(8)))  int v8i;
typedef __attribute__((ext_vector_type(4)))  int v4i;

#define WMMA_BF16(a, b, c) \
  __builtin_amdgcn_wmma_f32_16x16x32_bf16(false, (a), false, (b), (short)0, (c), false, false)

// ---------------------------------------------------------------------------
// CDNA5 async copy: GLOBAL_LOAD_ASYNC_TO_LDS_B128 (ASYNCcnt-tracked, writes
// LDS directly, no VGPR round trip, no stall at issue).
// ---------------------------------------------------------------------------
__device__ __forceinline__ void async_ld_b128(void* lds, const void* gptr) {
  unsigned loff = (unsigned)(size_t)lds;  // generic->LDS: low 32 bits = offset
  asm volatile("global_load_async_to_lds_b128 %0, %1, off"
               :: "v"(loff), "v"(gptr) : "memory");
}
#define WAIT_ASYNC(n) asm volatile("s_wait_asynccnt " #n ::: "memory")

// ---------------------------------------------------------------------------
// CDNA5 Tensor Data Mover: one instruction DMAs a whole 2D tile into LDS.
// Descriptor per CDNA5 ISA (D# groups 0/1), data_size=2B.
// This toolchain exposes the 6-arg builtin: (v4u, v8i, v4i, v4i, v8i, cpol).
// ---------------------------------------------------------------------------
#if __has_builtin(__builtin_amdgcn_tensor_load_to_lds)
#define HAVE_TDM 1
__device__ __forceinline__ void tdm_load_2d(void* lds_dst, const void* gsrc,
                                            unsigned tile_d0, unsigned tile_d1,
                                            unsigned tensor_d0, unsigned tensor_d1,
                                            unsigned stride_elems) {
  unsigned lds_addr = (unsigned)(size_t)lds_dst;
  unsigned long long ga = (unsigned long long)(size_t)gsrc;
  v4u g0 = { 1u,                                     // count=1, user descriptor
             lds_addr,                               // lds_addr [63:32]
             (unsigned)(ga & 0xFFFFFFFFu),           // global_addr[31:0]
             (unsigned)((ga >> 32) & 0x01FFFFFFu) | (2u << 30) }; // addr[56:32], type=2
  v8i g1 = { (int)(1u << 16),                        // data_size=1 (2 bytes)
             (int)((tensor_d0 & 0xFFFFu) << 16),     // tensor_dim0[15:0]
             (int)((tensor_d0 >> 16) | ((tensor_d1 & 0xFFFFu) << 16)),
             (int)((tensor_d1 >> 16) | (tile_d0 << 16)),  // tile_dim0
             (int)tile_d1,                           // tile_dim1 (tile_dim2=0)
             (int)stride_elems,                      // tensor_dim0_stride[31:0]
             0, 0 };
  v4i gz4 = {0, 0, 0, 0};
  v8i gz8 = {0, 0, 0, 0, 0, 0, 0, 0};
  __builtin_amdgcn_tensor_load_to_lds(g0, g1, gz4, gz4, gz8, 0);
}
#else
#define HAVE_TDM 0
#endif

// ---------------------------------------------------------------------------
// Fragment loader for 16-bit A (16xK slab, row=M) and B (KxN slab, lane=N
// holding the N-th column == N-th row of the row-major source).
// ISA layout: lanes 0-15 -> index lane, K chunks {0..7, 16..23};
//             lanes 16-31 -> index lane-16, K chunks {8..15, 24..31}.
// ---------------------------------------------------------------------------
__device__ __forceinline__ v16bf load_frag(const bf16* base, int ld) {
  const int lane = threadIdx.x & 31;
  const int r    = lane & 15;
  const int koff = (lane & 16) ? 8 : 0;
  const bf16* p = base + r * ld + koff;
  v8bf lo = *(const v8bf*)(p);
  v8bf hi = *(const v8bf*)(p + 16);
  v16bf f;
#pragma unroll
  for (int i = 0; i < 8; ++i) { f[i] = lo[i]; f[i + 8] = hi[i]; }
  return f;
}

// ---------------------------------------------------------------------------
// Precision / layout conversion kernels
// ---------------------------------------------------------------------------
__global__ void cvt_f32_bf16(const float* __restrict__ in, bf16* __restrict__ out, int n) {
  int i = blockIdx.x * blockDim.x + threadIdx.x;
  if (i < n) out[i] = (bf16)in[i];
}

// X [8][512][1024] (ch-major) -> Xbf [8][1024][512] (token-major)
__global__ void xpose_bf16(const float* __restrict__ X, bf16* __restrict__ Xbf) {
  int i = blockIdx.x * blockDim.x + threadIdx.x;
  if (i < 8 * 1024 * 512) {
    int c = i & 511;
    int n = (i >> 9) & 1023;
    int b = i >> 19;
    Xbf[i] = (bf16)X[((size_t)b * 512 + c) * 1024 + n];
  }
}

// ---------------------------------------------------------------------------
// GEMM: C[M,N] = A[M,K] * W[N,K]^T (+ epilogue). Block tile 128x128, 8 waves
// as 2x4, wave tile 64x32 -> 8 WMMA per 32-deep K step. Double-buffered LDS
// filled with async-to-LDS copies: tile k+1 streams while tile k computes.
// ---------------------------------------------------------------------------
#define BM 128
#define BN 128
#define BK 32

// Per-thread staging: 128x32 bf16 tile = 512 x 16B chunks = 2 chunks/thread.
#define STAGE_TILES(buf, kb)                                                      \
  do {                                                                            \
    async_ld_b128(&As[buf][ra0 * BK + ka0], &A[(size_t)(bm + ra0) * K + (kb) + ka0]); \
    async_ld_b128(&As[buf][ra1 * BK + ka1], &A[(size_t)(bm + ra1) * K + (kb) + ka1]); \
    async_ld_b128(&Bs[buf][ra0 * BK + ka0], &W[(size_t)(bn + ra0) * K + (kb) + ka0]); \
    async_ld_b128(&Bs[buf][ra1 * BK + ka1], &W[(size_t)(bn + ra1) * K + (kb) + ka1]); \
  } while (0)

#define GEMM_BODY()                                                    \
  const int bm = blockIdx.x * BM, bn = blockIdx.y * BN;                \
  const int tid = threadIdx.x, lane = tid & 31, wave = tid >> 5;       \
  const int wm = (wave & 1) * 64, wn = (wave >> 1) * 32;               \
  const int ra0 = tid >> 2, ka0 = (tid & 3) * 8;                       \
  const int ra1 = (tid + 256) >> 2, ka1 = ka0;                         \
  v8f acc[4][2] = {{{}, {}}, {{}, {}}, {{}, {}}, {{}, {}}};            \
  const int nk = K / BK;                                               \
  STAGE_TILES(0, 0);                                                   \
  for (int ki = 0; ki < nk; ++ki) {                                    \
    const int cur = ki & 1;                                            \
    if (ki + 1 < nk) {                                                 \
      STAGE_TILES(cur ^ 1, (size_t)(ki + 1) * BK);                     \
      WAIT_ASYNC(0x4); /* tile ki landed; next tile still in flight */ \
    } else {                                                           \
      WAIT_ASYNC(0x0);                                                 \
    }                                                                  \
    __syncthreads();                                                   \
    v16bf b0 = load_frag(&Bs[cur][(wn + 0) * BK], BK);                 \
    v16bf b1 = load_frag(&Bs[cur][(wn + 16) * BK], BK);                \
    _Pragma("unroll")                                                  \
    for (int mi = 0; mi < 4; ++mi) {                                   \
      v16bf a = load_frag(&As[cur][(wm + 16 * mi) * BK], BK);          \
      acc[mi][0] = WMMA_BF16(a, b0, acc[mi][0]);                       \
      acc[mi][1] = WMMA_BF16(a, b1, acc[mi][1]);                       \
    }                                                                  \
    __syncthreads(); /* protect the buffer the next stage overwrites */\
  }                                                                    \
  const int coln = lane & 15;                                          \
  const int rbase = (lane & 16) ? 8 : 0;

__global__ void gemm_qkv(const bf16* __restrict__ A, const bf16* __restrict__ W,
                         const float* __restrict__ bias, bf16* __restrict__ C,
                         int M, int N, int K) {
  __shared__ bf16 As[2][BM * BK];
  __shared__ bf16 Bs[2][BN * BK];
  GEMM_BODY()
#pragma unroll
  for (int mi = 0; mi < 4; ++mi)
#pragma unroll
    for (int ni = 0; ni < 2; ++ni) {
      int n0 = bn + wn + ni * 16 + coln;
      float bv = bias[n0];
#pragma unroll
      for (int r = 0; r < 8; ++r) {
        int m = bm + wm + mi * 16 + rbase + r;
        C[(size_t)m * N + n0] = (bf16)(acc[mi][ni][r] + bv);
      }
    }
}

__global__ void gemm_mlp(const bf16* __restrict__ A, const bf16* __restrict__ W,
                         const float* __restrict__ bias, const float* __restrict__ X,
                         float* __restrict__ out, int M, int N, int K) {
  __shared__ bf16 As[2][BM * BK];
  __shared__ bf16 Bs[2][BN * BK];
  GEMM_BODY()
#pragma unroll
  for (int mi = 0; mi < 4; ++mi)
#pragma unroll
    for (int ni = 0; ni < 2; ++ni) {
      int c = bn + wn + ni * 16 + coln;
      float bv = bias[c];
#pragma unroll
      for (int r = 0; r < 8; ++r) {
        int mrow = bm + wm + mi * 16 + rbase + r;
        int batch = mrow >> 10, n = mrow & 1023;
        size_t idx = ((size_t)(batch * 512 + c)) * 1024 + n;
        out[idx] = acc[mi][ni][r] + bv + X[idx];  // residual add, back to NCHW
      }
    }
}

// ---------------------------------------------------------------------------
// Attention: one block per (batch, head). K and V^T for the head live fully
// in LDS (2 * 128KB of the 320KB WGP LDS). The K block is DMA'd by the
// Tensor Data Mover in a single tensor_load_to_lds (TENSORcnt) while all
// waves overlap the V transpose fill. 8 waves then run flash-style online
// softmax over 16-query tiles against 32-key blocks.
// ---------------------------------------------------------------------------
#define ATTN_LDS ((1024 * 64 + 64 * 1024 + 8 * 16 * 32) * sizeof(bf16))

__global__ void attn_kernel(const bf16* __restrict__ qkv, bf16* __restrict__ scores) {
  extern __shared__ bf16 smem[];
  bf16* Ks = smem;                 // [1024 keys][64 dims]
  bf16* Vt = smem + 1024 * 64;     // [64 dims][1024 keys]  (transposed for B frags)
  bf16* Ps = smem + 2 * 1024 * 64; // per-wave 16x32 P transpose scratch

  const int bh = blockIdx.x;
  const int b = bh >> 3, h = bh & 7;
  const size_t rowbase = (size_t)b * 1024;
  const int tid = threadIdx.x, lane = tid & 31, wave = tid >> 5;

#if HAVE_TDM
  // Single TDM descriptor: 1024x64 bf16 tile, row stride 1536 elements.
  if (wave == 0)
    tdm_load_2d(Ks, qkv + rowbase * 1536 + h * 192 + 64,
                /*tile*/ 64, 1024, /*tensor*/ 64, 1024, /*stride*/ 1536);
#else
  for (int t = tid; t < 1024; t += 256) {
    const bf16* kp = qkv + (rowbase + t) * 1536 + h * 192 + 64;
#pragma unroll
    for (int d = 0; d < 64; d += 8)
      async_ld_b128(&Ks[t * 64 + d], kp + d);
  }
#endif

  // V transpose fill overlaps the K DMA: vectorized global reads, b16 scatter.
  for (int t = tid; t < 1024; t += 256) {
    const bf16* vp = qkv + (rowbase + t) * 1536 + h * 192 + 128;
#pragma unroll
    for (int d8 = 0; d8 < 64; d8 += 8) {
      v8bf vv = *(const v8bf*)(vp + d8);
#pragma unroll
      for (int i = 0; i < 8; ++i)
        Vt[(d8 + i) * 1024 + t] = vv[i];
    }
  }

#if HAVE_TDM
  if (wave == 0) __builtin_amdgcn_s_wait_tensorcnt(0);
#else
  WAIT_ASYNC(0x0);
#endif
  __syncthreads();

  bf16* Pw = Ps + wave * (16 * 32);
  const int coln  = lane & 15;
  const int rbase = (lane & 16) ? 8 : 0;

  for (int qt = wave; qt < 64; qt += 8) {
    const int q0 = qt * 16;
    const bf16* qbase = qkv + (rowbase + q0) * 1536 + h * 192;
    v16bf a0 = load_frag(qbase, 1536);      // q dims 0..31
    v16bf a1 = load_frag(qbase + 32, 1536); // q dims 32..63

    float m[8], l[8], scale[8];
    v8f o[4] = {{}, {}, {}, {}};
#pragma unroll
    for (int r = 0; r < 8; ++r) { m[r] = -1e30f; l[r] = 0.f; }

    for (int j0 = 0; j0 < 1024; j0 += 32) {
      // S = Q K^T for 16 queries x 32 keys
      v8f s0 = {}, s1 = {};
      s0 = WMMA_BF16(a0, load_frag(&Ks[j0 * 64], 64), s0);
      s0 = WMMA_BF16(a1, load_frag(&Ks[j0 * 64 + 32], 64), s0);
      s1 = WMMA_BF16(a0, load_frag(&Ks[(j0 + 16) * 64], 64), s1);
      s1 = WMMA_BF16(a1, load_frag(&Ks[(j0 + 16) * 64 + 32], 64), s1);
#pragma unroll
      for (int r = 0; r < 8; ++r) { s0[r] *= 0.125f; s1[r] *= 0.125f; }

      // Online softmax: row stats reduced across the 16 lanes of each half.
#pragma unroll
      for (int r = 0; r < 8; ++r) {
        float v = fmaxf(s0[r], s1[r]);
#pragma unroll
        for (int msk = 8; msk; msk >>= 1) v = fmaxf(v, __shfl_xor(v, msk, 32));
        float mn = fmaxf(m[r], v);
        scale[r] = __expf(m[r] - mn);
        m[r] = mn;
        float p0 = __expf(s0[r] - mn);
        float p1 = __expf(s1[r] - mn);
        s0[r] = p0; s1[r] = p1;
        float rs = p0 + p1;
#pragma unroll
        for (int msk = 8; msk; msk >>= 1) rs += __shfl_xor(rs, msk, 32);
        l[r] = l[r] * scale[r] + rs;
      }
#pragma unroll
      for (int d = 0; d < 4; ++d)
#pragma unroll
        for (int r = 0; r < 8; ++r) o[d][r] *= scale[r];

      // Transpose P (C layout -> A layout) through per-wave LDS scratch.
#pragma unroll
      for (int r = 0; r < 8; ++r) {
        Pw[(rbase + r) * 32 + coln]      = (bf16)s0[r];
        Pw[(rbase + r) * 32 + 16 + coln] = (bf16)s1[r];
      }
      v16bf pa = load_frag(Pw, 32);
#pragma unroll
      for (int d = 0; d < 4; ++d)
        o[d] = WMMA_BF16(pa, load_frag(&Vt[(d * 16) * 1024 + j0], 1024), o[d]);
    }

    // Normalize and store head-concat scores (bf16) for the MLP GEMM.
#pragma unroll
    for (int r = 0; r < 8; ++r) {
      float inv = 1.0f / l[r];
      size_t row = rowbase + q0 + rbase + r;
#pragma unroll
      for (int d = 0; d < 4; ++d)
        scores[row * 512 + h * 64 + d * 16 + coln] = (bf16)(o[d][r] * inv);
    }
  }
}

// ---------------------------------------------------------------------------
extern "C" void kernel_launch(void* const* d_in, const int* in_sizes, int n_in,
                              void* d_out, int out_size, void* d_ws, size_t ws_size,
                              hipStream_t stream) {
  const float* X    = (const float*)d_in[0];
  const float* Wprj = (const float*)d_in[1];
  const float* bprj = (const float*)d_in[2];
  const float* Wmlp = (const float*)d_in[3];
  const float* bmlp = (const float*)d_in[4];
  float* out = (float*)d_out;

  char* ws = (char*)d_ws;
  size_t off = 0;
  auto take = [&](size_t elems) {
    bf16* p = (bf16*)(ws + off);
    off = (off + elems * sizeof(bf16) + 255) & ~(size_t)255;
    return p;
  };
  bf16* Wprj_bf = take((size_t)1536 * 512);
  bf16* Wmlp_bf = take((size_t)512 * 512);
  bf16* Xbf     = take((size_t)8192 * 512);
  bf16* qkv     = take((size_t)8192 * 1536);
  bf16* sc      = take((size_t)8192 * 512);

  cvt_f32_bf16<<<(1536 * 512 + 255) / 256, 256, 0, stream>>>(Wprj, Wprj_bf, 1536 * 512);
  cvt_f32_bf16<<<(512 * 512 + 255) / 256, 256, 0, stream>>>(Wmlp, Wmlp_bf, 512 * 512);
  xpose_bf16<<<(8 * 1024 * 512 + 255) / 256, 256, 0, stream>>>(X, Xbf);

  gemm_qkv<<<dim3(8192 / BM, 1536 / BN), 256, 0, stream>>>(
      Xbf, Wprj_bf, bprj, qkv, 8192, 1536, 512);

  attn_kernel<<<64, 256, ATTN_LDS, stream>>>(qkv, sc);

  gemm_mlp<<<dim3(8192 / BM, 512 / BN), 256, 0, stream>>>(
      sc, Wmlp_bf, bmlp, X, out, 8192, 512, 512);
}
